// GNCC_19404662243724
// MI455X (gfx1250) — matrix operand order repaired and checked by reading, not compiled, not run on hardware
//
#include <hip/hip_runtime.h>

typedef _Float16 half_t;
typedef __attribute__((ext_vector_type(16))) _Float16 v16h;
typedef __attribute__((ext_vector_type(8)))  _Float16 v8h;
typedef __attribute__((ext_vector_type(8)))  float    v8f;

#define IN_CH 16
#define HID 16
#define EDGE_DIM 16
#define NCLS 40
#define FDIM 256           // IN_CH*HID
#define KT 8               // 256 / 32 k-tiles (layer 2)
#define NT 16              // 256 / 16 n-tiles
#define WPB 8              // waves per block (256 threads)
#define ET 4               // edge tiles (16 edges each) per wave: B-fragment reuse

// ---------------------------------------------------------------------------
// Workspace prep: zero accumulators, convert weights to f16 B-fragment layout.
// B-fragment (16x16x32 f16): lane = N column (lane&15), lanes 0-15 hold
// K=0..7,16..23 (e<8 -> K=e, else K=8+e), lanes 16-31 hold K=8..15,24..31 (e<8 -> K=8+e, else K=16+e).
// Linear storage: frag_base*512 + lane*16 + e   (halves)
// ---------------------------------------------------------------------------
__global__ void zero_f32_kernel(float* p, int n) {
    int i = blockIdx.x * blockDim.x + threadIdx.x;
    for (; i < n; i += gridDim.x * blockDim.x) p[i] = 0.0f;
}

__global__ void prep_w1_kernel(const float* __restrict__ w1, half_t* __restrict__ w1f) {
    int t = blockIdx.x * blockDim.x + threadIdx.x;
    if (t >= NT * 512) return;
    int n = t >> 9;                 // n-tile
    int r = t & 511;
    int lane = r >> 4, e = r & 15;
    int K = (e & 7) + ((lane >> 4) << 3) + ((e >> 3) << 4); // 0..31
    int col = n * 16 + (lane & 15);
    float v = (K < EDGE_DIM) ? w1[K * FDIM + col] : 0.0f;   // zero-pad K 16..31
    w1f[t] = (half_t)v;
}

__global__ void prep_w2_kernel(const float* __restrict__ w2, half_t* __restrict__ w2f) {
    int t = blockIdx.x * blockDim.x + threadIdx.x;
    if (t >= KT * NT * 512) return;
    int f = t >> 9;                 // f = k*16 + n
    int k = f >> 4, n = f & 15;
    int r = t & 511;
    int lane = r >> 4, e = r & 15;
    int K = 32 * k + (e & 7) + ((lane >> 4) << 3) + ((e >> 3) << 4);
    int col = n * 16 + (lane & 15);
    w2f[t] = (half_t)w2[K * FDIM + col];
}

// ---------------------------------------------------------------------------
// Helpers for the fused edge pipeline
// ---------------------------------------------------------------------------
__device__ __forceinline__ v16h make_attr_frag(const float* __restrict__ eattr,
                                               int e0t, int E, int hi, int lcol) {
    // A layout (16x32 f16): lanes 0-15: M=lane, e=0..7 -> K=e; lanes 16-31:
    // M=lane-16, e=0..7 -> K=8+e; e=8..15 -> K>=16 -> zero pad (K 16 -> 32).
    int erow = e0t + lcol;
    bool rv = erow < E;
    const float* ap = eattr + (long)(rv ? erow : (E - 1)) * EDGE_DIM + hi * 8;
    v16h a;
    #pragma unroll
    for (int e = 0; e < 8; ++e) a[e] = rv ? (half_t)ap[e] : (half_t)0.0f;
    #pragma unroll
    for (int e = 8; e < 16; ++e) a[e] = (half_t)0.0f;
    return a;
}

// Layer 1 for one 16-edge tile: h1 = relu(attr@w1 + b1), staged to LDS as f16,
// then reloaded as 8 A-fragments (LDS performs the C->A transpose).
__device__ __forceinline__ void edge_layer1(v16h a1,
                                            const half_t* __restrict__ w1f,
                                            const float* __restrict__ b1,
                                            half_t* hrow, v16h* a2,
                                            int lane, int hi, int lcol) {
    #pragma unroll
    for (int n = 0; n < NT; ++n) {
        v16h bw = *(const v16h*)(w1f + n * 512 + lane * 16);
        v8f acc = {};
        acc = __builtin_amdgcn_wmma_f32_16x16x32_f16(false, a1, false, bw,
                                                     (short)0, acc, false, false);
        float bb = b1[n * 16 + lcol];
        #pragma unroll
        for (int r = 0; r < 8; ++r) {
            float hv = acc[r] + bb;            // C: row M=r+8*hi, col N=lcol
            hv = hv > 0.0f ? hv : 0.0f;
            hrow[(r + hi * 8) * FDIM + n * 16 + lcol] = (half_t)hv;
        }
    }
    asm volatile("" ::: "memory");  // LDS store->load order (DS in-order per wave)
    #pragma unroll
    for (int k = 0; k < KT; ++k) {
        const half_t* p0 = hrow + lcol * FDIM + 32 * k + hi * 8;
        v8h lo = *(const v8h*)p0;          // K = 32k + 8*hi + 0..7
        v8h up = *(const v8h*)(p0 + 16);   // K = 32k + 16 + 8*hi + 0..7
        #pragma unroll
        for (int e = 0; e < 8; ++e) { a2[k][e] = lo[e]; a2[k][e + 8] = up[e]; }
    }
    asm volatile("" ::: "memory");  // guard staging WAR before next tile reuses it
}

// ---------------------------------------------------------------------------
// Fused edge pipeline: edge-MLP (two f16 WMMA GEMM layers, f32 accumulate),
// per-edge matvec msg = x[src] . W_e, atomic scatter-add into agg / cnt.
// Each wave handles ET*16 edges; each loaded w2 B-fragment feeds ET WMMAs.
// ---------------------------------------------------------------------------
__launch_bounds__(256)
__global__ void edge_kernel(const float* __restrict__ x,
                            const int*   __restrict__ srcIdx,
                            const int*   __restrict__ dstIdx,
                            const float* __restrict__ eattr,
                            const float* __restrict__ b1,
                            const float* __restrict__ b2,
                            const half_t* __restrict__ w1f,
                            const half_t* __restrict__ w2f,
                            float* __restrict__ agg,
                            float* __restrict__ cnt,
                            int E) {
    __shared__ half_t s_h1[WPB][16 * FDIM];        // 8 KB per wave staging
    __shared__ float  s_x [WPB][ET * 16 * IN_CH];  // gathered x[src] rows
    __shared__ int    s_dst[WPB][ET * 16];

    const int lane = threadIdx.x & 31;
    const int wave = threadIdx.x >> 5;
    const int hi   = lane >> 4;        // lane-half: 0 or 1
    const int lcol = lane & 15;

    const int tile = blockIdx.x * WPB + wave;
    const int e0   = tile * (ET * 16);
    if (e0 >= E) return;               // wave-uniform exit

    // --- gather src rows / dst ids (all 32 lanes, ET/2 edges each) ----------
    #pragma unroll
    for (int g = 0; g < (ET * 16) / 32; ++g) {
        int eidx  = e0 + g * 32 + lane;
        bool v    = eidx < E;
        int ec    = v ? eidx : (E - 1);
        int s = srcIdx[ec];
        int d = dstIdx[ec];
        s_dst[wave][g * 32 + lane] = v ? d : -1;
        const float* xr = x + (long)s * IN_CH;
        #pragma unroll
        for (int i = 0; i < IN_CH; ++i)
            s_x[wave][(g * 32 + lane) * IN_CH + i] = v ? xr[i] : 0.0f;
        if (v) atomicAdd(&cnt[d], 1.0f);
    }

    // --- layer 1 for all edge tiles (staging buffer reused sequentially) ----
    half_t* hrow = s_h1[wave];
    v16h a2[ET][KT];
    #pragma unroll
    for (int t = 0; t < ET; ++t) {
        v16h a1 = make_attr_frag(eattr, e0 + t * 16, E, hi, lcol);
        edge_layer1(a1, w1f, b1, hrow, a2[t], lane, hi, lcol);
    }

    // --- layer 2 + fused per-edge matvec; each bw fragment feeds ET WMMAs ---
    // Column tile n of h2 is exactly W[:, i=n, o=0..15]; accumulate
    // msg[e,o] += x[src[e], n] * h2[e,16n+o] with lane-local FMAs.
    v8f msg[ET];
    #pragma unroll
    for (int t = 0; t < ET; ++t) msg[t] = (v8f){};
    const float* xs = s_x[wave];
    #pragma unroll
    for (int n = 0; n < NT; ++n) {
        v8f acc[ET];
        #pragma unroll
        for (int t = 0; t < ET; ++t) acc[t] = (v8f){};
        #pragma unroll
        for (int k = 0; k < KT; ++k) {
            v16h bw = *(const v16h*)(w2f + (size_t)(k * 16 + n) * 512 + lane * 16);
            #pragma unroll
            for (int t = 0; t < ET; ++t)
                acc[t] = __builtin_amdgcn_wmma_f32_16x16x32_f16(false, a2[t][k], false, bw,
                                                                (short)0, acc[t], false, false);
        }
        float b2v = b2[n * 16 + lcol];
        #pragma unroll
        for (int t = 0; t < ET; ++t) {
            #pragma unroll
            for (int r = 0; r < 8; ++r) {
                int row = r + hi * 8;
                msg[t][r] += xs[(t * 16 + row) * IN_CH + n] * (acc[t][r] + b2v);
            }
        }
    }

    // --- scatter-add into destination nodes ---------------------------------
    #pragma unroll
    for (int t = 0; t < ET; ++t) {
        #pragma unroll
        for (int r = 0; r < 8; ++r) {
            int row = r + hi * 8;
            int d = s_dst[wave][t * 16 + row];
            if (d >= 0) atomicAdd(&agg[(long)d * HID + lcol], msg[t][r]);
        }
    }
}

// ---------------------------------------------------------------------------
// Node head: out = relu(agg/max(cnt,1) + x@root + bias); classifier MLP.
// Memory-bound; one thread per node.
// ---------------------------------------------------------------------------
__global__ void node_kernel(const float* __restrict__ x,
                            const float* __restrict__ agg,
                            const float* __restrict__ cnt,
                            const float* __restrict__ root,
                            const float* __restrict__ bias,
                            const float* __restrict__ wc1,
                            const float* __restrict__ bc1,
                            const float* __restrict__ wc2,
                            const float* __restrict__ bc2,
                            float* __restrict__ out, int Nn) {
    int n = blockIdx.x * blockDim.x + threadIdx.x;
    if (n >= Nn) return;
    float c = cnt[n]; c = c > 1.0f ? c : 1.0f;
    float inv = 1.0f / c;

    float xr[IN_CH];
    #pragma unroll
    for (int i = 0; i < IN_CH; ++i) xr[i] = x[(long)n * IN_CH + i];

    float t[HID];
    #pragma unroll
    for (int o = 0; o < HID; ++o) t[o] = agg[(long)n * HID + o] * inv + bias[o];
    for (int i = 0; i < IN_CH; ++i) {
        float xv = xr[i];
        #pragma unroll
        for (int o = 0; o < HID; ++o) t[o] += xv * root[i * HID + o];
    }
    #pragma unroll
    for (int o = 0; o < HID; ++o) t[o] = t[o] > 0.0f ? t[o] : 0.0f;

    float u[HID];
    #pragma unroll
    for (int j = 0; j < HID; ++j) u[j] = bc1[j];
    for (int i = 0; i < HID; ++i) {
        float tv = t[i];
        #pragma unroll
        for (int j = 0; j < HID; ++j) u[j] += tv * wc1[i * HID + j];
    }
    #pragma unroll
    for (int j = 0; j < HID; ++j) u[j] = u[j] > 0.0f ? u[j] : 0.0f;

    for (int k = 0; k < NCLS; ++k) {
        float s = bc2[k];
        #pragma unroll
        for (int i = 0; i < HID; ++i) s += u[i] * wc2[i * NCLS + k];
        out[(long)n * NCLS + k] = s;
    }
}

// ---------------------------------------------------------------------------
extern "C" void kernel_launch(void* const* d_in, const int* in_sizes, int n_in,
                              void* d_out, int out_size, void* d_ws, size_t ws_size,
                              hipStream_t stream) {
    const float* x     = (const float*)d_in[0];
    const int*   eidx  = (const int*)  d_in[1];
    const float* eattr = (const float*)d_in[2];
    const float* w1    = (const float*)d_in[3];
    const float* b1    = (const float*)d_in[4];
    const float* w2    = (const float*)d_in[5];
    const float* b2    = (const float*)d_in[6];
    const float* root  = (const float*)d_in[7];
    const float* bias  = (const float*)d_in[8];
    const float* wc1   = (const float*)d_in[9];
    const float* bc1   = (const float*)d_in[10];
    const float* wc2   = (const float*)d_in[11];
    const float* bc2   = (const float*)d_in[12];
    float* out = (float*)d_out;

    const int Nn = in_sizes[0] / IN_CH;
    const int E  = in_sizes[1] / 2;

    // workspace layout
    char* ws = (char*)d_ws;
    size_t off = 0;
    float* agg = (float*)(ws + off); off += (size_t)Nn * HID * sizeof(float);
    float* cnb = (float*)(ws + off); off += (size_t)Nn * sizeof(float);
    off = (off + 255) & ~(size_t)255;
    half_t* w1f = (half_t*)(ws + off); off += (size_t)NT * 512 * sizeof(half_t);
    off = (off + 255) & ~(size_t)255;
    half_t* w2f = (half_t*)(ws + off); off += (size_t)KT * NT * 512 * sizeof(half_t);

    zero_f32_kernel<<<512, 256, 0, stream>>>(agg, Nn * HID);
    zero_f32_kernel<<<128, 256, 0, stream>>>(cnb, Nn);
    prep_w1_kernel<<<(NT * 512 + 255) / 256, 256, 0, stream>>>(w1, w1f);
    prep_w2_kernel<<<(KT * NT * 512 + 255) / 256, 256, 0, stream>>>(w2, w2f);

    const int tiles  = (E + ET * 16 - 1) / (ET * 16);
    const int blocks = (tiles + WPB - 1) / WPB;
    edge_kernel<<<blocks, 32 * WPB, 0, stream>>>(x, eidx, eidx + E, eattr,
                                                 b1, b2, w1f, w2f, agg, cnb, E);

    node_kernel<<<(Nn + 255) / 256, 256, 0, stream>>>(x, agg, cnb, root, bias,
                                                      wc1, bc1, wc2, bc2, out, Nn);
}
